// AdvancedFeatureExtractor_39865886441542
// MI455X (gfx1250) — compile-verified, matrix operand order checked by reading, not run on hardware
//
#include <hip/hip_runtime.h>
#include <hip/hip_bf16.h>
#include <math.h>

typedef __attribute__((ext_vector_type(16))) _Float16 v16h;
typedef __attribute__((ext_vector_type(8)))  float    v8f;
typedef __attribute__((ext_vector_type(4)))  unsigned int u32x4;
typedef __attribute__((ext_vector_type(8)))  int          i32x8;
typedef __attribute__((ext_vector_type(4)))  int          i32x4;

constexpr int B      = 64;
constexpr int L      = 4094;
constexpr int M      = B * L;          // 262016 fused positions (divisible by 16)
constexpr int C      = 28;             // channels per conv branch
constexpr int KCONV  = 84;             // 3*28 conv channels -> fusion rows 0..83
constexpr int KPAD   = 96;             // padded to 3 K-tiles of 32
constexpr int DHID   = 85;             // small-MLP hidden dim
constexpr int NH     = 256;            // fusion hidden / output dim
constexpr int NBINS  = 2048;           // rfft bins (L/2+1)
constexpr int CONST_K = 283;           // fusion rows 84..366 (constant per b)
constexpr int NTILES_M = M / 16;       // 16376
constexpr unsigned A_TILE_BYTES = 16 * KPAD * 2;  // 3072 B per A tile

// ---- workspace layout (bytes, all offsets multiples of 32) ----
constexpr size_t O_CONVRAW  = 0;
constexpr size_t O_CONVF16  = O_CONVRAW  + (size_t)KCONV * M * 4;   // 88.0 MB
constexpr size_t O_MAG      = O_CONVF16  + (size_t)M * KPAD * 2;    // 50.3 MB
constexpr size_t O_PH       = O_MAG      + (size_t)B * NBINS * 4;
constexpr size_t O_BANDMAG  = O_PH       + (size_t)B * NBINS * 4;
constexpr size_t O_BANDPH   = O_BANDMAG  + (size_t)B * 8 * 4;
constexpr size_t O_SF       = O_BANDPH   + (size_t)B * 8 * 4;
constexpr size_t O_FF       = O_SF       + (size_t)B * 28 * 4;
constexpr size_t O_WAVFEAT  = O_FF       + (size_t)B * 170 * 4;
constexpr size_t O_WF       = O_WAVFEAT  + (size_t)B * 8 * 4;
constexpr size_t O_CHANSCALE= O_WF       + (size_t)B * 85 * 4;
constexpr size_t O_C1       = O_CHANSCALE+ 1024;
constexpr size_t O_W1FRAG   = O_C1       + (size_t)B * NH * 4;
constexpr size_t O_W2FRAG   = O_W1FRAG   + (size_t)3 * 16 * 512 * 2;
// total ~139.8 MB

__constant__ float DB4LOr[8] = { 0.23037781330885523f, 0.7148465705525415f,
    0.6308807679295904f, -0.02798376941698385f, -0.18703481171888114f,
    0.030841381835986965f, 0.032883011666982945f, -0.010597401784997278f };
__constant__ float DB4HIr[8] = { -0.010597401784997278f, 0.032883011666982945f,
    0.030841381835986965f, 0.18703481171888114f, -0.02798376941698385f,
    -0.6308807679295904f, 0.7148465705525415f, -0.23037781330885523f };

__device__ inline float blockReduce(float* red, float v, int tid) {
  red[tid] = v; __syncthreads();
  for (int s = 128; s > 0; s >>= 1) {
    if (tid < s) red[tid] += red[tid + s];
    __syncthreads();
  }
  float r = red[0]; __syncthreads();
  return r;
}

// ---- Tensor Data Mover: 1-D global->LDS DMA (D# per CDNA5 ISA) ----
__device__ inline void tdmLoad1D(unsigned ldsAddr, const void* gsrc, unsigned nBytes) {
  unsigned long long ga = (unsigned long long)(size_t)gsrc;
  unsigned n8 = nBytes >> 3;                    // 8-byte elements
  u32x4 g0;
  g0[0] = 1u;                                   // count=1 (valid user descriptor)
  g0[1] = ldsAddr;                              // lds_addr (bytes)
  g0[2] = (unsigned)ga;                         // global_addr[31:0]
  g0[3] = (unsigned)(ga >> 32) | (2u << 30);    // global_addr[56:32] | type=2 (image)
  i32x8 g1;
  g1[0] = (int)(3u << 16);                      // workgroup_mask=0, data_size=3 (8 B)
  g1[1] = (int)((n8 & 0xFFFFu) << 16);          // tensor_dim0[15:0]  (bits 63:48)
  g1[2] = (int)(n8 >> 16);                      // tensor_dim0[31:16]; tensor_dim1=0
  g1[3] = (int)((n8 & 0xFFFFu) << 16);          // tile_dim0 (bits 127:112)
  g1[4] = 0;                                    // tile_dim1/2 = 0 -> 1-D
  g1[5] = (int)n8;                              // tensor_dim0_stride low32
  g1[6] = 0;
  g1[7] = 0;
  i32x4 gz = {0, 0, 0, 0};
#if defined(__clang_major__) && (__clang_major__ >= 23)
  i32x8 gz8 = {0, 0, 0, 0, 0, 0, 0, 0};
  __builtin_amdgcn_tensor_load_to_lds(g0, g1, gz, gz, gz8, 0);
#else
  __builtin_amdgcn_tensor_load_to_lds(g0, g1, gz, gz, 0);
#endif
}

// ---------- per-row stats -> st linear ----------
__global__ void statsK(const float* __restrict__ x, const float* __restrict__ st_w,
                       const float* __restrict__ st_b, float* __restrict__ sf) {
  __shared__ float s1[256], s2[256], mx[256], mn[256], st[4];
  int b = blockIdx.x, tid = threadIdx.x;
  const float* xr = x + (size_t)b * L;
  float a = 0.f, q = 0.f, h = -INFINITY, l = INFINITY;
  for (int i = tid; i < L; i += 256) {
    float v = xr[i]; a += v; q = fmaf(v, v, q);
    h = fmaxf(h, v); l = fminf(l, v);
  }
  s1[tid] = a; s2[tid] = q; mx[tid] = h; mn[tid] = l; __syncthreads();
  for (int s = 128; s > 0; s >>= 1) {
    if (tid < s) { s1[tid] += s1[tid + s]; s2[tid] += s2[tid + s];
      mx[tid] = fmaxf(mx[tid], mx[tid + s]); mn[tid] = fminf(mn[tid], mn[tid + s]); }
    __syncthreads();
  }
  if (tid == 0) {
    float mean = s1[0] / (float)L;
    float var = (s2[0] - (float)L * mean * mean) / (float)(L - 1); // ddof=1
    st[0] = mean; st[1] = sqrtf(fmaxf(var, 0.f)); st[2] = mx[0]; st[3] = mn[0];
  }
  __syncthreads();
  if (tid < C) {
    float o = st_b[tid];
    for (int i = 0; i < 4; ++i) o = fmaf(st[i], st_w[i * C + tid], o);
    sf[b * C + tid] = o;
  }
}

// ---------- temporal convolutions (raw, pre-batchnorm) ----------
template<int K>
__device__ inline void convWin(const float* xs, const float* wsh, const float* bias,
                               float* convRaw, int cbase, int bIdx, int t, int tid, bool valid) {
  const int x0 = tid + 64 - K / 2;
  for (int c = 0; c < C; ++c) {
    float acc = bias[c];
    const float* wc = wsh + c * K;
    #pragma unroll 8
    for (int j = 0; j < K; ++j) acc = fmaf(wc[j], xs[x0 + j], acc);
    if (valid) convRaw[(size_t)(cbase + c) * M + (size_t)bIdx * L + t] = acc;
  }
}

__global__ void convK(const float* __restrict__ x,
                      const float* __restrict__ w0, const float* __restrict__ b0,
                      const float* __restrict__ w1, const float* __restrict__ b1,
                      const float* __restrict__ w2, const float* __restrict__ b2,
                      float* __restrict__ convRaw) {
  __shared__ float xs[384];     // 256 tile + 64 left halo + 63 right halo
  __shared__ float wsh[6272];   // all conv weights
  int bIdx = blockIdx.x >> 4;
  int t0 = (blockIdx.x & 15) * 256;
  int tid = threadIdx.x;
  for (int i = tid; i < 896;  i += 256) wsh[i]        = w0[i];
  for (int i = tid; i < 1792; i += 256) wsh[896 + i]  = w1[i];
  for (int i = tid; i < 3584; i += 256) wsh[2688 + i] = w2[i];
  for (int i = tid; i < 384; i += 256) {
    int g = t0 - 64 + i;
    xs[i] = (g >= 0 && g < L) ? x[(size_t)bIdx * L + g] : 0.f;
  }
  __syncthreads();
  int t = t0 + tid; bool valid = t < L;
  convWin<32 >(xs, wsh,        b0, convRaw, 0,  bIdx, t, tid, valid);
  convWin<64 >(xs, wsh + 896,  b1, convRaw, 28, bIdx, t, tid, valid);
  convWin<128>(xs, wsh + 2688, b2, convRaw, 56, bIdx, t, tid, valid);
}

// ---------- per-channel BN stats -> scale/shift (deterministic tree reduce) ----------
__global__ void chanReduceK(const float* __restrict__ convRaw,
                            const float* g0, const float* bt0,
                            const float* g1, const float* bt1,
                            const float* g2, const float* bt2,
                            float* __restrict__ chanScale) {
  __shared__ float red[256];
  int c = blockIdx.x, tid = threadIdx.x;
  const float* p = convRaw + (size_t)c * M;
  float s = 0.f, q = 0.f;
  for (int i = tid; i < M; i += 256) { float v = p[i]; s += v; q = fmaf(v, v, q); }
  s = blockReduce(red, s, tid);
  q = blockReduce(red, q, tid);
  if (tid == 0) {
    float mean = s / (float)M;
    float var = q / (float)M - mean * mean;   // ddof=0
    float g, bt;
    if (c < 28)      { g = g0[c];      bt = bt0[c]; }
    else if (c < 56) { g = g1[c - 28]; bt = bt1[c - 28]; }
    else             { g = g2[c - 56]; bt = bt2[c - 56]; }
    float sc = g / sqrtf(var + 1e-5f);
    chanScale[2 * c] = sc;
    chanScale[2 * c + 1] = bt - mean * sc;
  }
}

// ---------- normalize + ReLU + pack conv feats to f16 [pos][96] ----------
__global__ void packConvK(const float* __restrict__ convRaw,
                          const float* __restrict__ chanScale,
                          _Float16* __restrict__ convF16) {
  int pos = blockIdx.x * 256 + threadIdx.x;
  if (pos >= M) return;
  _Float16* dst = convF16 + (size_t)pos * KPAD;
  for (int c = 0; c < KCONV; ++c) {
    float v = fmaf(convRaw[(size_t)c * M + pos], chanScale[2 * c], chanScale[2 * c + 1]);
    dst[c] = (_Float16)fmaxf(v, 0.f);
  }
  for (int c = KCONV; c < KPAD; ++c) dst[c] = (_Float16)0.f;
}

// ---------- DFT via incremental rotation (resync per 512-sample chunk) ----------
__global__ void dftK(const float* __restrict__ x, float* __restrict__ mag,
                     float* __restrict__ ph) {
  __shared__ float xs[512];
  int b = blockIdx.x >> 3;
  int f = (blockIdx.x & 7) * 256 + threadIdx.x;
  const float w = -2.0f * 3.14159265358979323846f * (float)f / (float)L;
  float dsn, dc; __sincosf(w, &dsn, &dc);
  float re = 0.f, im = 0.f;
  for (int t0 = 0; t0 < L; t0 += 512) {
    __syncthreads();
    for (int i = threadIdx.x; i < 512; i += 256) {
      int t = t0 + i;
      xs[i] = (t < L) ? x[(size_t)b * L + t] : 0.f;
    }
    __syncthreads();
    float s, c; __sincosf(w * (float)t0, &s, &c);  // exact resync each chunk
    int lim = min(512, L - t0);
    for (int i = 0; i < lim; ++i) {
      float xv = xs[i];
      re = fmaf(xv, c, re);
      im = fmaf(xv, s, im);
      float nc = c * dc - s * dsn;
      s = fmaf(c, dsn, s * dc);
      c = nc;
    }
  }
  mag[(size_t)b * NBINS + f] = sqrtf(re * re + im * im);
  ph [(size_t)b * NBINS + f] = atan2f(im, re);
}

// ---------- band means ----------
__global__ void bandK(const float* __restrict__ mag, const float* __restrict__ ph,
                      float* __restrict__ bandMag, float* __restrict__ bandPh) {
  int b = blockIdx.x, tid = threadIdx.x;
  if (tid < 8) {
    float s = 0.f; const float* p = mag + (size_t)b * NBINS + tid * 256;
    for (int i = 0; i < 256; ++i) s += p[i];
    bandMag[b * 8 + tid] = s / 256.f;
  } else if (tid < 16) {
    int bd = tid - 8;
    float s = 0.f; const float* p = ph + (size_t)b * NBINS + bd * 256;
    for (int i = 0; i < 256; ++i) s += p[i];
    bandPh[b * 8 + bd] = s / 256.f;
  }
}

// ---------- tiny MLP 8 -> 85 -> 85 ----------
__global__ void mlpK(const float* __restrict__ in, const float* __restrict__ w1,
                     const float* __restrict__ b1, const float* __restrict__ w2,
                     const float* __restrict__ b2, float* __restrict__ out,
                     int outPitch, int outOff) {
  __shared__ float h1[DHID];
  __shared__ float xin[8];
  int b = blockIdx.x, tid = threadIdx.x;
  if (tid < 8) xin[tid] = in[b * 8 + tid];
  __syncthreads();
  if (tid < DHID) {
    float a = b1[tid];
    for (int i = 0; i < 8; ++i) a = fmaf(xin[i], w1[i * DHID + tid], a);
    h1[tid] = fmaxf(a, 0.f);
  }
  __syncthreads();
  if (tid < DHID) {
    float a = b2[tid];
    for (int i = 0; i < DHID; ++i) a = fmaf(h1[i], w2[i * DHID + tid], a);
    out[(size_t)b * outPitch + outOff + tid] = a;
  }
}

// ---------- 4-level DB4 wavelet: |cD| mean + std per level ----------
__global__ void waveletK(const float* __restrict__ x, float* __restrict__ wavFeat) {
  __shared__ float bufA[4094];
  __shared__ float bufB[2051];
  __shared__ float red[256];
  int b = blockIdx.x, tid = threadIdx.x;
  for (int i = tid; i < L; i += 256) bufA[i] = x[(size_t)b * L + i];
  __syncthreads();
  int n = L;
  for (int lvl = 0; lvl < 4; ++lvl) {
    int nOut = (n + 6) / 2 + 1;
    float s1 = 0.f, s2 = 0.f, sa = 0.f;
    for (int o = tid; o < nOut; o += 256) {
      float a = 0.f, d = 0.f;
      #pragma unroll
      for (int j = 0; j < 8; ++j) {
        int i = 2 * o + j - 7;           // symmetric pad of 7 each side
        if (i < 0)  i = -1 - i;
        if (i >= n) i = 2 * n - 1 - i;
        float v = bufA[i];
        a = fmaf(DB4LOr[j], v, a);
        d = fmaf(DB4HIr[j], v, d);
      }
      bufB[o] = a;
      s1 += d; s2 = fmaf(d, d, s2); sa += fabsf(d);
    }
    s1 = blockReduce(red, s1, tid);
    s2 = blockReduce(red, s2, tid);
    sa = blockReduce(red, sa, tid);
    if (tid == 0) {
      float mean = s1 / (float)nOut;
      float var = s2 / (float)nOut - mean * mean;   // ddof=0
      wavFeat[b * 8 + (3 - lvl) * 2]     = sa / (float)nOut;  // pairs[::-1]
      wavFeat[b * 8 + (3 - lvl) * 2 + 1] = sqrtf(fmaxf(var, 0.f));
    }
    __syncthreads();
    for (int i = tid; i < nOut; i += 256) bufA[i] = bufB[i];
    __syncthreads();
    n = nOut;
  }
}

// ---------- fold constant channels: c1[b,n] = const_feats[b] @ W1[84:] + b1 ----------
__global__ void c1K(const float* __restrict__ sf, const float* __restrict__ ff,
                    const float* __restrict__ wf, const float* __restrict__ fu_w1,
                    const float* __restrict__ fu_b1, float* __restrict__ c1) {
  int b = blockIdx.x, n = threadIdx.x;
  float acc = fu_b1[n];
  for (int r = 0; r < CONST_K; ++r) {
    float cv = (r < 28)  ? sf[b * 28 + r]
             : (r < 198) ? ff[b * 170 + (r - 28)]
                         : wf[b * 85 + (r - 198)];
    acc = fmaf(cv, fu_w1[(size_t)(84 + r) * NH + n], acc);
  }
  c1[b * NH + n] = acc;
}

// ---------- pack weights into pre-swizzled WMMA B-fragments (ISA 16x32 f16 layout) ----------
__device__ inline void fragMap(int lp, int& kl, int& nl) {
  int lane = lp >> 4, slot = lp & 15;
  int hsel = lane >> 4; nl = lane & 15;
  int v = slot >> 1, pos = slot & 1;
  kl = ((v < 4) ? 2 * v : 16 + 2 * (v - 4)) + 8 * hsel + pos;
}

__global__ void packW1K(const float* __restrict__ fu_w1, _Float16* __restrict__ w1frag) {
  int idx = blockIdx.x * 256 + threadIdx.x;
  if (idx >= 3 * 16 * 512) return;
  int frag = idx >> 9, lp = idx & 511;
  int kt = frag >> 4, nt = frag & 15;
  int kl, nl; fragMap(lp, kl, nl);
  int k = kt * 32 + kl, nn = nt * 16 + nl;
  float v = (k < KCONV) ? fu_w1[(size_t)k * NH + nn] : 0.f;  // rows 0..83, zero-pad 84..95
  w1frag[idx] = (_Float16)v;
}

__global__ void packW2K(const float* __restrict__ fu_w2, _Float16* __restrict__ w2frag) {
  int idx = blockIdx.x * 256 + threadIdx.x;    // 8*16*512 = 65536 total
  int frag = idx >> 9, lp = idx & 511;
  int kt = frag >> 4, nt = frag & 15;
  int kl, nl; fragMap(lp, kl, nl);
  w2frag[idx] = (_Float16)fu_w2[(size_t)(kt * 32 + kl) * NH + nt * 16 + nl];
}

// ---------- fused per-position MLP: WMMA f16 GEMMs, TDM double-buffered A tiles ----------
__global__ void __launch_bounds__(256)
fusionK(const _Float16* __restrict__ convF16, const _Float16* __restrict__ w1fragG,
        const _Float16* __restrict__ w2fragG, const float* __restrict__ c1,
        const float* __restrict__ fu_b2, float* __restrict__ out) {
  __shared__ __attribute__((aligned(32))) _Float16 w1f[3 * 16 * 512]; // 48 KB, resident
  __shared__ __attribute__((aligned(16))) _Float16 aT0[16 * KPAD];    // 3 KB A ping
  __shared__ __attribute__((aligned(16))) _Float16 aT1[16 * KPAD];    // 3 KB A pong
  __shared__ __attribute__((aligned(16))) _Float16 hT[16 * NH];       // 8 KB hidden tile
  __shared__ float b2s[NH];
  int tid = threadIdx.x;
  {
    const uint4* g = (const uint4*)w1fragG;
    uint4* s = (uint4*)w1f;
    for (int i = tid; i < 3072; i += 256) s[i] = g[i];
    b2s[tid] = fu_b2[tid];
  }
  const int lane = tid & 31, wave = tid >> 5;
  const int m = lane & 15, hsel = lane >> 4, nl = lane & 15;
  const v16h* w1v = (const v16h*)w1f;
  const v16h* w2v = (const v16h*)w2fragG;
  const unsigned aAddr[2] = { (unsigned)(size_t)&aT0[0], (unsigned)(size_t)&aT1[0] };
  int kloc[8];
  #pragma unroll
  for (int v = 0; v < 8; ++v) kloc[v] = ((v < 4) ? 2 * v : 16 + 2 * (v - 4)) + 8 * hsel;

  // prologue: TDM DMA of the first A tile (wave 0 issues, EXEC-independent)
  if (wave == 0)
    tdmLoad1D(aAddr[0], convF16 + (size_t)blockIdx.x * 16 * KPAD, A_TILE_BYTES);
  int cur = 0;

  for (int tile = blockIdx.x; tile < NTILES_M; tile += gridDim.x) {
    const int mbase = tile * 16;
    const int bb0   = mbase / L;          // one scalar division per tile
    const int lbase = mbase - bb0 * L;    // rows cross at most one batch boundary
    const int nextTile = tile + (int)gridDim.x;
    if (wave == 0) {
      if (nextTile < NTILES_M) {
        tdmLoad1D(aAddr[cur ^ 1], convF16 + (size_t)nextTile * 16 * KPAD, A_TILE_BYTES);
        __builtin_amdgcn_s_wait_tensorcnt(1);  // current tile's DMA has landed
      } else {
        __builtin_amdgcn_s_wait_tensorcnt(0);
      }
    }
    __syncthreads();
    const _Float16* aT = cur ? aT1 : aT0;

    // GEMM1: H = ReLU(A(16x96) @ W1(96x256) + c1)
    #pragma unroll
    for (int nt2 = 0; nt2 < 2; ++nt2) {
      const int ntile = wave * 2 + nt2;
      const int n = ntile * 16 + nl;
      v8f acc;
      #pragma unroll
      for (int r = 0; r < 8; ++r) {
        int roff = r + 8 * hsel;
        int idx = bb0 * NH + n + ((lbase + roff >= L) ? NH : 0);
        acc[r] = c1[idx];
      }
      #pragma unroll
      for (int kt = 0; kt < 3; ++kt) {
        v16h a;
        #pragma unroll
        for (int v = 0; v < 8; ++v) {
          const _Float16* p = &aT[m * KPAD + kt * 32 + kloc[v]];
          a[2 * v] = p[0]; a[2 * v + 1] = p[1];
        }
        v16h bfr = w1v[(kt * 16 + ntile) * 32 + lane];
        acc = __builtin_amdgcn_wmma_f32_16x16x32_f16(false, a, false, bfr,
                                                     (short)0, acc, false, false);
      }
      #pragma unroll
      for (int r = 0; r < 8; ++r)
        hT[(r + 8 * hsel) * NH + ntile * 16 + nl] = (_Float16)fmaxf(acc[r], 0.f);
    }
    __syncthreads();

    // GEMM2: OUT = H(16x256) @ W2(256x256) + b2
    #pragma unroll
    for (int nt2 = 0; nt2 < 2; ++nt2) {
      const int ntile = wave * 2 + nt2;
      const int n = ntile * 16 + nl;
      v8f acc;
      #pragma unroll
      for (int r = 0; r < 8; ++r) acc[r] = b2s[n];
      #pragma unroll
      for (int kt = 0; kt < 8; ++kt) {
        v16h a;
        #pragma unroll
        for (int v = 0; v < 8; ++v) {
          const _Float16* p = &hT[m * NH + kt * 32 + kloc[v]];
          a[2 * v] = p[0]; a[2 * v + 1] = p[1];
        }
        v16h bfr = w2v[(kt * 16 + ntile) * 32 + lane];
        acc = __builtin_amdgcn_wmma_f32_16x16x32_f16(false, a, false, bfr,
                                                     (short)0, acc, false, false);
      }
      const size_t base = (size_t)bb0 * NH * L + (size_t)n * L + (size_t)lbase;
      #pragma unroll
      for (int r = 0; r < 8; ++r) {
        int roff = r + 8 * hsel;
        size_t off = base + roff + ((lbase + roff >= L) ? (size_t)(NH - 1) * L : 0);
        out[off] = acc[r];
      }
    }
    __syncthreads();
    cur ^= 1;
  }
}

extern "C" void kernel_launch(void* const* d_in, const int* in_sizes, int n_in,
                              void* d_out, int out_size, void* d_ws, size_t ws_size,
                              hipStream_t stream) {
  (void)in_sizes; (void)n_in; (void)out_size; (void)ws_size;
  const float* x      = (const float*)d_in[0];
  const float* tc_w0  = (const float*)d_in[1];
  const float* tc_b0  = (const float*)d_in[2];
  const float* bn_g0  = (const float*)d_in[3];
  const float* bn_b0  = (const float*)d_in[4];
  const float* tc_w1  = (const float*)d_in[5];
  const float* tc_b1  = (const float*)d_in[6];
  const float* bn_g1  = (const float*)d_in[7];
  const float* bn_b1  = (const float*)d_in[8];
  const float* tc_w2  = (const float*)d_in[9];
  const float* tc_b2  = (const float*)d_in[10];
  const float* bn_g2  = (const float*)d_in[11];
  const float* bn_b2  = (const float*)d_in[12];
  const float* st_w   = (const float*)d_in[13];
  const float* st_b   = (const float*)d_in[14];
  const float* mag_w1 = (const float*)d_in[15];
  const float* mag_b1 = (const float*)d_in[16];
  const float* mag_w2 = (const float*)d_in[17];
  const float* mag_b2 = (const float*)d_in[18];
  const float* ph_w1  = (const float*)d_in[19];
  const float* ph_b1  = (const float*)d_in[20];
  const float* ph_w2  = (const float*)d_in[21];
  const float* ph_b2  = (const float*)d_in[22];
  const float* wv_w1  = (const float*)d_in[23];
  const float* wv_b1  = (const float*)d_in[24];
  const float* wv_w2  = (const float*)d_in[25];
  const float* wv_b2  = (const float*)d_in[26];
  const float* fu_w1  = (const float*)d_in[27];
  const float* fu_b1  = (const float*)d_in[28];
  const float* fu_w2  = (const float*)d_in[29];
  const float* fu_b2  = (const float*)d_in[30];

  char* ws = (char*)d_ws;
  float*    convRaw   = (float*)(ws + O_CONVRAW);
  _Float16* convF16   = (_Float16*)(ws + O_CONVF16);
  float*    mag       = (float*)(ws + O_MAG);
  float*    ph        = (float*)(ws + O_PH);
  float*    bandMag   = (float*)(ws + O_BANDMAG);
  float*    bandPh    = (float*)(ws + O_BANDPH);
  float*    sf        = (float*)(ws + O_SF);
  float*    ff        = (float*)(ws + O_FF);
  float*    wavFeat   = (float*)(ws + O_WAVFEAT);
  float*    wf        = (float*)(ws + O_WF);
  float*    chanScale = (float*)(ws + O_CHANSCALE);
  float*    c1        = (float*)(ws + O_C1);
  _Float16* w1frag    = (_Float16*)(ws + O_W1FRAG);
  _Float16* w2frag    = (_Float16*)(ws + O_W2FRAG);
  float*    outF      = (float*)d_out;

  statsK     <<<B,       256, 0, stream>>>(x, st_w, st_b, sf);
  convK      <<<B * 16,  256, 0, stream>>>(x, tc_w0, tc_b0, tc_w1, tc_b1, tc_w2, tc_b2, convRaw);
  chanReduceK<<<KCONV,   256, 0, stream>>>(convRaw, bn_g0, bn_b0, bn_g1, bn_b1, bn_g2, bn_b2, chanScale);
  packConvK  <<<(M + 255) / 256, 256, 0, stream>>>(convRaw, chanScale, convF16);
  dftK       <<<B * 8,   256, 0, stream>>>(x, mag, ph);
  bandK      <<<B,       64,  0, stream>>>(mag, ph, bandMag, bandPh);
  mlpK       <<<B,       128, 0, stream>>>(bandMag, mag_w1, mag_b1, mag_w2, mag_b2, ff, 170, 0);
  mlpK       <<<B,       128, 0, stream>>>(bandPh,  ph_w1,  ph_b1,  ph_w2,  ph_b2,  ff, 170, 85);
  waveletK   <<<B,       256, 0, stream>>>(x, wavFeat);
  mlpK       <<<B,       128, 0, stream>>>(wavFeat, wv_w1, wv_b1, wv_w2, wv_b2, wf, 85, 0);
  c1K        <<<B,       256, 0, stream>>>(sf, ff, wf, fu_w1, fu_b1, c1);
  packW1K    <<<96,      256, 0, stream>>>(fu_w1, w1frag);
  packW2K    <<<256,     256, 0, stream>>>(fu_w2, w2frag);
  fusionK    <<<1024,    256, 0, stream>>>(convF16, w1frag, w2frag, c1, fu_b2, outF);
}
